// Model_50809463111716
// MI455X (gfx1250) — matrix operand order repaired and checked by reading, not compiled, not run on hardware
//
#include <hip/hip_runtime.h>

typedef __attribute__((ext_vector_type(16))) _Float16 v16h;
typedef __attribute__((ext_vector_type(8)))  _Float16 v8h;
typedef __attribute__((ext_vector_type(8)))  float    v8f;

// Model dims
#define TV   256
#define TE   256
#define TF   256
#define THE  256
#define THD  512
#define TP   128
#define TO   80
#define TOP  96      /* TO zero-padded to a multiple of 32 for the GEMM */
#define TK   10
#define TTIN 256
#define TTOUT 512
#define TB   32

// ---------------------------------------------------------------------------
__device__ __forceinline__ float sigm(float x){ return 1.0f/(1.0f+__expf(-x)); }

__device__ __forceinline__ float dropmask(unsigned idx, unsigned seed){
  unsigned h = idx*2654435761u ^ seed;
  h ^= h>>13; h *= 0x5bd1e995u; h ^= h>>15;
  return (h & 1u) ? 2.0f : 0.0f;   // keep=0.5, scale 1/keep
}

#define WMMA_F16(a,b,c) __builtin_amdgcn_wmma_f32_16x16x32_f16(false,(a),false,(b),(short)0,(c),false,false)

// ---------------------------------------------------------------------------
// One A-segment MAC sweep over a PAIR of row tiles sharing one B fragment.
// A f16 [M x Kseg] (ld=Kseg, Kseg % 32 == 0). Lane's 16 A halves = two
// contiguous 16B chunks (ISA 16-bit A layout: {k0+hi*8..+7},{k0+hi*8+16..+23}).
// B f16 transposed [N][Ktot]; lane's 16 halves = same two 16B chunks as A.
// ---------------------------------------------------------------------------
__device__ __forceinline__ void seg_mac2(const _Float16* __restrict__ p, int Kseg,
                                         int& kg, int row0, int row1, int n, int hi,
                                         const _Float16* __restrict__ Wt, int Ktot,
                                         v8f& acc0, v8f& acc1)
{
  if (Kseg <= 0) return;
  const _Float16* a0p = p + (size_t)row0*Kseg + hi*8;
  const _Float16* a1p = p + (size_t)row1*Kseg + hi*8;
  const _Float16* bp  = Wt + (size_t)n*Ktot + kg + hi*8;
  for (int k0 = 0; k0 + 32 <= Kseg; k0 += 32) {
    v8h a0lo = *(const v8h*)(a0p + k0);
    v8h a0hi = *(const v8h*)(a0p + k0 + 16);
    v8h a1lo = *(const v8h*)(a1p + k0);
    v8h a1hi = *(const v8h*)(a1p + k0 + 16);
    v16h b   = __builtin_shufflevector(*(const v8h*)bp, *(const v8h*)(bp + 16), 0,1,2,3,4,5,6,7,8,9,10,11,12,13,14,15);
    __builtin_prefetch(bp + 32, 0, 1);               // global_prefetch_b8
    bp += 32;
    v16h a0 = __builtin_shufflevector(a0lo, a0hi, 0,1,2,3,4,5,6,7,8,9,10,11,12,13,14,15);
    v16h a1 = __builtin_shufflevector(a1lo, a1hi, 0,1,2,3,4,5,6,7,8,9,10,11,12,13,14,15);
    acc0 = WMMA_F16(a0, b, acc0);
    acc1 = WMMA_F16(a1, b, acc1);
  }
  kg += Kseg;
}

// ---------------------------------------------------------------------------
// Generic WMMA GEMM: C[M,N] = concat(A0..A3)[M,Ktot] @ W[Ktot,N] + bias.
// M % 32 == 0; every Kseg % 32 == 0. One wave per (row-tile-pair, col-tile):
// B loaded once, used by both row tiles (halves L2 weight traffic).
// ---------------------------------------------------------------------------
__global__ void k_gemm(const _Float16* s0,int K0,const _Float16* s1,int K1,
                       const _Float16* s2,int K2,const _Float16* s3,int K3,
                       const _Float16* __restrict__ Wt, const float* __restrict__ bias,
                       float* __restrict__ C, int M, int N)
{
  const int Ktot = K0+K1+K2+K3;
  const int lane = threadIdx.x & 31;
  const int tid  = blockIdx.x*8 + (threadIdx.x >> 5);
  const int ntiles = N >> 4;
  const int pairs  = M >> 5;
  if (tid >= pairs*ntiles) return;
  const int pr = tid / ntiles, nt = tid % ntiles;
  const int row0 = pr*32 + (lane & 15);
  const int row1 = row0 + 16;
  const int n    = nt*16 + (lane & 15);
  const int hi   = lane >> 4;

  v8f acc0 = {}, acc1 = {};
  int kg = 0;
  seg_mac2(s0, K0, kg, row0, row1, n, hi, Wt, Ktot, acc0, acc1);
  seg_mac2(s1, K1, kg, row0, row1, n, hi, Wt, Ktot, acc0, acc1);
  seg_mac2(s2, K2, kg, row0, row1, n, hi, Wt, Ktot, acc0, acc1);
  seg_mac2(s3, K3, kg, row0, row1, n, hi, Wt, Ktot, acc0, acc1);

  const float bv = bias ? bias[n] : 0.0f;
  const int rbase = pr*32 + hi*8;
#pragma unroll
  for (int r = 0; r < 8; ++r) {
    C[(size_t)(rbase + r)*N + n]      = acc0[r] + bv;
    C[(size_t)(rbase + 16 + r)*N + n] = acc1[r] + bv;
  }
}

// ---------------------------------------------------------------------------
// Implicit-GEMM conv, same row-tile pairing. pin: padded masked f16 input
// [B][T+4][F]. K dt-major: k = dt*256 + i (32-wide k-tile never crosses dt).
// Row pair differs only by t+16 (same batch). WcT: [F(out)][5*256] f16.
// ---------------------------------------------------------------------------
__global__ void k_conv(const _Float16* __restrict__ pin,
                       const _Float16* __restrict__ WcT,
                       const float* __restrict__ bias,
                       float* __restrict__ accout)
{
  const int Kt = 5*TF, N = TF, M = TB*TTIN;
  const int lane = threadIdx.x & 31;
  const int tid  = blockIdx.x*8 + (threadIdx.x >> 5);
  const int ntiles = N >> 4;
  const int pairs  = M >> 5;
  if (tid >= pairs*ntiles) return;
  const int pr = tid / ntiles, nt = tid % ntiles;
  const int row0 = pr*32 + (lane & 15);
  const int b = row0 >> 8, t0 = row0 & 255;    // row1 -> t0+16, same b
  const int n = nt*16 + (lane & 15);
  const int hi = lane >> 4;

  v8f acc0 = {}, acc1 = {};
  const _Float16* bp = WcT + (size_t)n*Kt + hi*8;
  for (int k0 = 0; k0 < Kt; k0 += 32) {
    const int dt = k0 >> 8;
    const int i0 = (k0 & 255) + hi*8;
    const _Float16* ap = pin + ((size_t)(b*(TTIN+4) + t0 + dt)*TF + i0);
    v8h a0lo = *(const v8h*)ap;
    v8h a0hi = *(const v8h*)(ap + 16);
    v8h a1lo = *(const v8h*)(ap + 16*TF);
    v8h a1hi = *(const v8h*)(ap + 16*TF + 16);
    v16h bf  = __builtin_shufflevector(*(const v8h*)(bp + k0), *(const v8h*)(bp + k0 + 16), 0,1,2,3,4,5,6,7,8,9,10,11,12,13,14,15);
    v16h a0 = __builtin_shufflevector(a0lo, a0hi, 0,1,2,3,4,5,6,7,8,9,10,11,12,13,14,15);
    v16h a1 = __builtin_shufflevector(a1lo, a1hi, 0,1,2,3,4,5,6,7,8,9,10,11,12,13,14,15);
    acc0 = WMMA_F16(a0, bf, acc0);
    acc1 = WMMA_F16(a1, bf, acc1);
  }
  const float bv = bias[n];
  const int rbase = pr*32 + hi*8;
#pragma unroll
  for (int r = 0; r < 8; ++r) {
    accout[(size_t)(rbase + r)*N + n]      = acc0[r] + bv;
    accout[(size_t)(rbase + 16 + r)*N + n] = acc1[r] + bv;
  }
}

// ---------------------------------------------------------------------------
// Small kernels
// ---------------------------------------------------------------------------
__global__ void k_embed(const int* __restrict__ x, const float* __restrict__ emb,
                        _Float16* __restrict__ e16, float* __restrict__ hbuf)
{
  int idx = blockIdx.x*blockDim.x + threadIdx.x;      // T*B*E
  if (idx >= TTIN*TB*TE) return;
  int ei = idx & 255, tb = idx >> 8;
  int b = tb & 31, t = tb >> 5;
  float v = emb[x[t*TB + b]*TE + ei];
  e16[idx] = (_Float16)v;
  hbuf[((size_t)b*TE + ei)*TTIN + t] = v;
}

// W f32 [Kd x N] -> Wt f16 [N x Kpad], zero pad K in [Kd, Kpad)
__global__ void k_wprep(const float* __restrict__ W, _Float16* __restrict__ Wt,
                        int Kd, int Kpad, int N)
{
  int idx = blockIdx.x*blockDim.x + threadIdx.x;      // N*Kpad
  if (idx >= N*Kpad) return;
  int k = idx % Kpad, n = idx / Kpad;
  Wt[idx] = (k < Kd) ? (_Float16)W[(size_t)k*N + n] : (_Float16)0.0f;
}

// Conv weights combined (k=1/3/5 -> padded k=5), dt-major K: WcT[o][dt*256+i]
__global__ void k_convprep(const float* __restrict__ w1, const float* __restrict__ w3,
                           const float* __restrict__ w5, _Float16* __restrict__ WcT)
{
  int idx = blockIdx.x*blockDim.x + threadIdx.x;      // F*5*F, (o, dt, i)
  if (idx >= TF*5*TF) return;
  int i  = idx & 255;
  int dt = (idx >> 8) % 5;
  int o  = idx / (5*256);
  float v = w5[((size_t)o*TF + i)*5 + dt];
  if (dt >= 1 && dt <= 3) v += w3[((size_t)o*TF + i)*3 + dt - 1];
  if (dt == 2)            v += w1[(size_t)o*TF + i];
  WcT[idx] = (_Float16)v;
}

__global__ void k_cb3(const float* a, const float* b, const float* c, float* o)
{ int i = threadIdx.x; if (i < TF) o[i] = a[i]+b[i]+c[i]; }

__global__ void k_zero(float* p, int n)
{ int i = blockIdx.x*blockDim.x + threadIdx.x; if (i < n) p[i] = 0.0f; }

__global__ void k_zero16(_Float16* p, int n)
{ int i = blockIdx.x*blockDim.x + threadIdx.x; if (i < n) p[i] = (_Float16)0.0f; }

__global__ void k_cvt(const float* x, _Float16* y, int n)
{ int i = blockIdx.x*blockDim.x + threadIdx.x; if (i < n) y[i] = (_Float16)x[i]; }

// Build padded, mask-premultiplied conv input: pin[b][t+2][i] (f16)
__global__ void k_convpad(const float* __restrict__ hin, const float* __restrict__ mask,
                          _Float16* __restrict__ pin)
{
  int idx = blockIdx.x*blockDim.x + threadIdx.x;      // B*(T+4)*F
  if (idx >= TB*(TTIN+4)*TF) return;
  int i = idx & 255, rem = idx >> 8;
  int tp = rem % (TTIN+4), b = rem / (TTIN+4);
  int t = tp - 2;
  float v = 0.0f;
  if ((unsigned)t < (unsigned)TTIN)
    v = hin[((size_t)b*TF + i)*TTIN + t] * mask[t*TB + b];
  pin[idx] = (_Float16)v;
}

// BN stats: one block per channel, LDS tree reduction over B*T rows.
__global__ void k_bnstats(const float* __restrict__ acc, float* __restrict__ mv)
{
  __shared__ float s1[256], s2[256];
  const int c = blockIdx.x, Mr = TB*TTIN;
  float su = 0.f, sq = 0.f;
  for (int r = threadIdx.x; r < Mr; r += 256) {
    float v = acc[(size_t)r*TF + c];
    su += v; sq += v*v;
  }
  s1[threadIdx.x] = su; s2[threadIdx.x] = sq;
  __syncthreads();
  for (int s = 128; s > 0; s >>= 1) {
    if (threadIdx.x < s) { s1[threadIdx.x]+=s1[threadIdx.x+s]; s2[threadIdx.x]+=s2[threadIdx.x+s]; }
    __syncthreads();
  }
  if (threadIdx.x == 0) {
    float mu = s1[0]/Mr;
    mv[c] = mu; mv[TF + c] = s2[0]/Mr - mu*mu;
  }
}

__global__ void k_bnapply(const float* __restrict__ acc, const float* __restrict__ mv,
                          const float* __restrict__ g, const float* __restrict__ bb,
                          const float* __restrict__ hin, const float* __restrict__ mask,
                          float* __restrict__ hout)
{
  int idx = blockIdx.x*blockDim.x + threadIdx.x;      // B*F*T
  if (idx >= TB*TF*TTIN) return;
  int t = idx & 255, rem = idx >> 8;
  int f = rem & 255, b = rem >> 8;
  float a  = acc[((size_t)b*TTIN + t)*TF + f];
  float nrm = (a - mv[f]) * rsqrtf(mv[TF+f] + 1e-5f) * g[f] + bb[f];
  float inp = hin[((size_t)b*TF + f)*TTIN + t] * mask[t*TB + b];
  hout[((size_t)b*TF + f)*TTIN + t] = fmaxf(nrm, 0.0f) + inp;
}

// masked transpose -> f16 encoder conv features (T,B,F)
__global__ void k_mtrans(const float* __restrict__ hbuf, const float* __restrict__ mask,
                         _Float16* __restrict__ c16)
{
  int idx = blockIdx.x*blockDim.x + threadIdx.x;      // T*B*F
  if (idx >= TTIN*TB*TF) return;
  int f = idx & 255, tb = idx >> 8;
  int b = tb & 31, t = tb >> 5;
  c16[idx] = (_Float16)(hbuf[((size_t)b*TF + f)*TTIN + t] * mask[t*TB + b]);
}

// LSTM gates + state update + masked blend. Writes f32 state, f16 state copy,
// optional f32 seq slices and f16 seq slice (GEMM A inputs for later stages).
__global__ void k_gates(const float* __restrict__ z, float* __restrict__ hs,
                        float* __restrict__ cs, _Float16* __restrict__ hs16,
                        const float* __restrict__ mask_t,
                        float* __restrict__ hseq, float* __restrict__ cseq,
                        _Float16* __restrict__ hseq16, int H)
{
  int idx = blockIdx.x*blockDim.x + threadIdx.x;
  if (idx >= TB*H) return;
  int b = idx / H, j = idx - b*H;
  const float* zb = z + (size_t)b*4*H;
  float ig = sigm(zb[j]);
  float fg = sigm(zb[H + j]);
  float gg = tanhf(zb[2*H + j]);
  float og = sigm(zb[3*H + j]);
  float cp = cs[idx], hp = hs[idx];
  float cn = fg*cp + ig*gg;
  float hn = og*tanhf(cn);
  float m = mask_t[b];
  hn = m*hn + (1.0f-m)*hp;
  cn = m*cn + (1.0f-m)*cp;
  hs[idx] = hn; cs[idx] = cn; hs16[idx] = (_Float16)hn;
  if (hseq)   hseq[idx]   = hn;
  if (cseq)   cseq[idx]   = cn;
  if (hseq16) hseq16[idx] = (_Float16)hn;
}

// dropout on y, written into K-padded f16 rows [TOUT*B][TOP]
__global__ void k_drop16pad(const float* __restrict__ x, _Float16* __restrict__ yp,
                            unsigned seed)
{
  int idx = blockIdx.x*blockDim.x + threadIdx.x;      // TOUT*B*TOP
  if (idx >= TTOUT*TB*TOP) return;
  int c = idx % TOP, row = idx / TOP;
  float v = 0.0f;
  if (c < TO) v = x[(size_t)row*TO + c] * dropmask(row*TO + c, seed);
  yp[idx] = (_Float16)v;
}

__global__ void k_relu_drop16(const float* x, _Float16* y, int n, unsigned seed)
{ int i = blockIdx.x*blockDim.x + threadIdx.x; if (i < n) y[i] = (_Float16)(fmaxf(x[i],0.f)*dropmask(i, seed)); }

__global__ void k_relu_cvt(const float* x, _Float16* y, int n)
{ int i = blockIdx.x*blockDim.x + threadIdx.x; if (i < n) y[i] = (_Float16)fmaxf(x[i], 0.f); }

// attention mixture params:  abk = hn @ Wk + bk  (32x30, K=512), then a/b/kappa
__global__ void k_abk(const float* __restrict__ hn, const float* __restrict__ Wk,
                      const float* __restrict__ bk, float* __restrict__ kstate,
                      const float* __restrict__ mask_t, float* __restrict__ ak_out,
                      float* __restrict__ a_s, float* __restrict__ b_s,
                      float* __restrict__ kraw_s)
{
  __shared__ float zsh[TB*3*TK];
  int tid = threadIdx.x;
  if (tid < TB*3*TK) {
    int b = tid/(3*TK), j = tid%(3*TK);
    float s = bk[j];
    const float* h = hn + (size_t)b*THD;
    for (int d = 0; d < THD; ++d) s += h[d]*Wk[d*(3*TK) + j];
    zsh[tid] = s;
  }
  __syncthreads();
  if (tid < TB*TK) {
    int b = tid/TK, j = tid%TK;
    float za = zsh[b*3*TK + j], zb = zsh[b*3*TK + TK + j], zk = zsh[b*3*TK + 2*TK + j];
    float a  = __expf(za), bc = __expf(zb);
    float sp = (zk > 20.f) ? zk : log1pf(__expf(zk));
    float kp = kstate[tid];
    float kr = kp + sp;
    a_s[tid] = a; b_s[tid] = bc; kraw_s[tid] = kr;
    float m = mask_t[b];
    float kn = m*kr + (1.0f-m)*kp;
    kstate[tid] = kn; ak_out[tid] = kn;
  }
}

__global__ void k_phi(const float* __restrict__ a_s, const float* __restrict__ b_s,
                      const float* __restrict__ kraw, const float* __restrict__ encmask,
                      const float* __restrict__ decmask_t,
                      float* __restrict__ phi, float* __restrict__ aphi_t)
{
  int idx = blockIdx.x*blockDim.x + threadIdx.x;      // B*TIN
  if (idx >= TB*TTIN) return;
  int b = idx / TTIN, ti = idx % TTIN;
  float s = 0.f;
#pragma unroll
  for (int k = 0; k < TK; ++k) {
    float d = kraw[b*TK + k] - (float)ti;
    s += a_s[b*TK + k] * __expf(-b_s[b*TK + k]*d*d);
  }
  float p = s * encmask[ti*TB + b];
  phi[idx] = p;
  aphi_t[idx] = p * decmask_t[b];
}

__global__ void k_ctx(const float* __restrict__ phi, const float* __restrict__ hfseq,
                      const float* __restrict__ hbseq, const float* __restrict__ decmask_t,
                      float* __restrict__ wstate, _Float16* __restrict__ w16,
                      float* __restrict__ aw_t, _Float16* __restrict__ aw16_t)
{
  int idx = blockIdx.x*blockDim.x + threadIdx.x;      // B * 2HE
  if (idx >= TB*2*THE) return;
  int b = idx / (2*THE), d = idx % (2*THE);
  const float* src = (d < THE) ? hfseq : hbseq;
  int dd = d & (THE-1);
  float s = 0.f;
  for (int t = 0; t < TTIN; ++t)
    s += phi[b*TTIN + t] * src[((size_t)t*TB + b)*THE + dd];
  float m = decmask_t[b];
  float wn = m*s + (1.0f-m)*wstate[idx];
  wstate[idx] = wn; w16[idx] = (_Float16)wn;
  aw_t[idx] = wn; aw16_t[idx] = (_Float16)wn;
}

// ---------------------------------------------------------------------------
extern "C" void kernel_launch(void* const* d_in, const int* in_sizes, int n_in,
                              void* d_out, int out_size, void* d_ws, size_t ws_size,
                              hipStream_t stream)
{
  (void)in_sizes; (void)n_in; (void)out_size; (void)ws_size;

  // ---- inputs (setup_inputs dict order, params dict flattened in insertion order)
  const int*   x       = (const int*)  d_in[0];
  const float* y       = (const float*)d_in[1];
  const float* encmask = (const float*)d_in[10];
  const float* decmask = (const float*)d_in[11];
  const float* emb     = (const float*)d_in[12];
  const float* cw[2][3] = {{(const float*)d_in[13],(const float*)d_in[15],(const float*)d_in[17]},
                           {(const float*)d_in[21],(const float*)d_in[23],(const float*)d_in[25]}};
  const float* cb[2][3] = {{(const float*)d_in[14],(const float*)d_in[16],(const float*)d_in[18]},
                           {(const float*)d_in[22],(const float*)d_in[24],(const float*)d_in[26]}};
  const float* bng[2] = {(const float*)d_in[19], (const float*)d_in[27]};
  const float* bnb[2] = {(const float*)d_in[20], (const float*)d_in[28]};
  const float* Wf = (const float*)d_in[29]; const float* bf = (const float*)d_in[30];
  const float* Wb = (const float*)d_in[31]; const float* bbk= (const float*)d_in[32];
  const float* Wp1= (const float*)d_in[33]; const float* bp1= (const float*)d_in[34];
  const float* Wp2= (const float*)d_in[35]; const float* bp2= (const float*)d_in[36];
  const float* Wat= (const float*)d_in[37]; const float* bat= (const float*)d_in[38];
  const float* Wk = (const float*)d_in[39]; const float* bk = (const float*)d_in[40];
  const float* W1 = (const float*)d_in[41]; const float* b1 = (const float*)d_in[42];
  const float* W2 = (const float*)d_in[43]; const float* b2 = (const float*)d_in[44];
  const float* Wo = (const float*)d_in[45]; const float* bo = (const float*)d_in[46];

  // ---- d_out slices (flat f32, return order)
  float* out = (float*)d_out;
  size_t off = 0;
  float* o_preds = out + off; off += (size_t)TTOUT*TB*TO;
  float* o_ah    = out + off; off += (size_t)TTOUT*TB*THD;
  float* o_ac    = out + off; off += (size_t)TTOUT*TB*THD;
  float* o_ak    = out + off; off += (size_t)TTOUT*TB*TK;
  float* o_aw    = out + off; off += (size_t)TTOUT*TB*2*THE;
  float* o_aphi  = out + off; off += (size_t)TTOUT*TB*TTIN;
  float* o_h1    = out + off; off += (size_t)TTOUT*TB*THD;
  float* o_c1    = out + off; off += (size_t)TTOUT*TB*THD;
  float* o_h2    = out + off; off += (size_t)TTOUT*TB*THD;
  float* o_c2    = out + off; off += (size_t)TTOUT*TB*THD;

  // ---- workspace carve
  char* wsb = (char*)d_ws; size_t woff = 0;
  auto alloc = [&](size_t bytes)->void* {
    void* p = wsb + woff; woff = (woff + bytes + 255) & ~(size_t)255; return p;
  };
  auto a16 = [&](size_t n){ return (_Float16*)alloc(n*sizeof(_Float16)); };
  auto a32 = [&](size_t n){ return (float*)alloc(n*sizeof(float)); };

  // weights (f16, transposed [N][K]) — ~25 MB total, L2-resident across scan
  _Float16* WtF  = a16((size_t)4*THE*(TE+TF+THE));
  _Float16* WtBw = a16((size_t)4*THE*(TE+TF+THE));
  _Float16* WtAt = a16((size_t)4*THD*(THE+2*THE+THD));
  _Float16* Wt1  = a16((size_t)4*THD*(THE+2*THE+2*THD));
  _Float16* Wt2  = a16((size_t)4*THD*(THE+2*THE+2*THD));
  _Float16* WtP1 = a16((size_t)TP*TOP);
  _Float16* WtP2 = a16((size_t)THE*TP);
  _Float16* WtO  = a16((size_t)TO*THD);   // N=80 is a multiple of 16 (5 col tiles)
  _Float16* WcT0 = a16((size_t)TF*5*TF);
  _Float16* WcT1 = a16((size_t)TF*5*TF);
  float* cbc0 = a32(TF); float* cbc1 = a32(TF);
  // f32 activation buffers
  float* hbufA= a32((size_t)TB*TF*TTIN);
  float* hbufB= a32((size_t)TB*TF*TTIN);
  float* accb = a32((size_t)TB*TTIN*TF);
  float* mv   = a32(2*TF);
  float* hfseq= a32((size_t)TTIN*TB*THE);
  float* hbseq= a32((size_t)TTIN*TB*THE);
  float* dp1  = a32((size_t)TTOUT*TB*TP);
  float* py2  = a32((size_t)TTOUT*TB*THE);
  float* zbuf = a32((size_t)TB*4*THD);
  float* hf_s = a32(TB*THE); float* cf_s = a32(TB*THE);
  float* hb_s = a32(TB*THE); float* cb_s = a32(TB*THE);
  float* ah_s = a32(TB*THD); float* ac_s = a32(TB*THD);
  float* w_s  = a32(TB*2*THE); float* kk_s = a32(TB*TK);
  float* h1_s = a32(TB*THD); float* c1_s = a32(TB*THD);
  float* h2_s = a32(TB*THD); float* c2_s = a32(TB*THD);
  float* a_sc = a32(TB*TK); float* b_sc = a32(TB*TK); float* kr_sc = a32(TB*TK);
  float* phib = a32(TB*TTIN);
  // f16 activation shadows (GEMM A inputs)
  _Float16* e16     = a16((size_t)TTIN*TB*TE);
  _Float16* c16     = a16((size_t)TTIN*TB*TF);
  _Float16* pin16   = a16((size_t)TB*(TTIN+4)*TF);
  _Float16* hf16    = a16(TB*THE);
  _Float16* hb16    = a16(TB*THE);
  _Float16* dy16    = a16((size_t)TTOUT*TB*TOP);
  _Float16* dp216   = a16((size_t)TTOUT*TB*TP);
  _Float16* py216   = a16((size_t)TTOUT*TB*THE);
  _Float16* ah16    = a16(TB*THD);
  _Float16* w16     = a16(TB*2*THE);
  _Float16* h116    = a16(TB*THD);
  _Float16* h216    = a16(TB*THD);
  _Float16* ahseq16 = a16((size_t)TTOUT*TB*THD);
  _Float16* awseq16 = a16((size_t)TTOUT*TB*2*THE);
  _Float16* h1seq16 = a16((size_t)TTOUT*TB*THD);
  _Float16* h2seq16 = a16((size_t)TTOUT*TB*THD);

  auto gemm = [&](const _Float16* s0,int K0,const _Float16* s1,int K1,
                  const _Float16* s2,int K2,const _Float16* s3,int K3,
                  const _Float16* Wt,const float* bias,float* C,int M,int N){
    int tiles = (M/32)*(N/16);
    k_gemm<<<(tiles+7)/8, 256, 0, stream>>>(s0,K0,s1,K1,s2,K2,s3,K3,Wt,bias,C,M,N);
  };
  auto gates = [&](const float* z, float* hs, float* cs, _Float16* hs16,
                   const float* mt, float* hseq, float* cq, _Float16* hseq16, int H){
    int n = TB*H;
    k_gates<<<(n+255)/256, 256, 0, stream>>>(z, hs, cs, hs16, mt, hseq, cq, hseq16, H);
  };
  auto wprep = [&](const float* W, _Float16* Wt, int Kd, int Kpad, int N){
    int n = N*Kpad;
    k_wprep<<<(n+255)/256, 256, 0, stream>>>(W, Wt, Kd, Kpad, N);
  };

  // ===== weight prep
  wprep(Wf,  WtF,  TE+TF+THE,       TE+TF+THE,       4*THE);
  wprep(Wb,  WtBw, TE+TF+THE,       TE+TF+THE,       4*THE);
  wprep(Wat, WtAt, THE+2*THE+THD,   THE+2*THE+THD,   4*THD);
  wprep(W1,  Wt1,  THE+2*THE+2*THD, THE+2*THE+2*THD, 4*THD);
  wprep(W2,  Wt2,  THE+2*THE+2*THD, THE+2*THE+2*THD, 4*THD);
  wprep(Wp1, WtP1, TO,  TOP, TP);
  wprep(Wp2, WtP2, TP,  TP,  THE);
  wprep(Wo,  WtO,  THD, THD, TO);
  k_convprep<<<(TF*5*TF+255)/256,256,0,stream>>>(cw[0][0],cw[0][1],cw[0][2],WcT0);
  k_convprep<<<(TF*5*TF+255)/256,256,0,stream>>>(cw[1][0],cw[1][1],cw[1][2],WcT1);
  k_cb3<<<1,256,0,stream>>>(cb[0][0],cb[0][1],cb[0][2],cbc0);
  k_cb3<<<1,256,0,stream>>>(cb[1][0],cb[1][1],cb[1][2],cbc1);

  // ===== embedding
  k_embed<<<(TTIN*TB*TE+255)/256,256,0,stream>>>(x, emb, e16, hbufA);

  // ===== conv stacks (padded-f16 implicit-GEMM WMMA + BN + ReLU + residual)
  const _Float16* WcT[2] = {WcT0, WcT1};
  const float* cbc[2] = {cbc0, cbc1};
  float* hin = hbufA; float* hout = hbufB;
  for (int s = 0; s < 2; ++s) {
    k_convpad<<<(TB*(TTIN+4)*TF+255)/256,256,0,stream>>>(hin, encmask, pin16);
    int tiles = (TB*TTIN/32)*(TF/16);
    k_conv<<<(tiles+7)/8,256,0,stream>>>(pin16, WcT[s], cbc[s], accb);
    k_bnstats<<<TF,256,0,stream>>>(accb, mv);
    k_bnapply<<<(TB*TF*TTIN+255)/256,256,0,stream>>>(accb, mv, bng[s], bnb[s],
                                                     hin, encmask, hout);
    float* tmp = hin; hin = hout; hout = tmp;
  }
  k_mtrans<<<(TTIN*TB*TF+255)/256,256,0,stream>>>(hin, encmask, c16);

  // ===== bidirectional encoder LSTM (256 sequential steps each way)
  k_zero<<<(TB*THE+255)/256,256,0,stream>>>(hf_s, TB*THE);
  k_zero<<<(TB*THE+255)/256,256,0,stream>>>(cf_s, TB*THE);
  k_zero<<<(TB*THE+255)/256,256,0,stream>>>(hb_s, TB*THE);
  k_zero<<<(TB*THE+255)/256,256,0,stream>>>(cb_s, TB*THE);
  k_zero16<<<(TB*THE+255)/256,256,0,stream>>>(hf16, TB*THE);
  k_zero16<<<(TB*THE+255)/256,256,0,stream>>>(hb16, TB*THE);
  for (int t = 0; t < TTIN; ++t) {
    gemm(e16 + (size_t)t*TB*TE, TE, c16 + (size_t)t*TB*TF, TF, hf16, THE,
         nullptr, 0, WtF, bf, zbuf, TB, 4*THE);
    gates(zbuf, hf_s, cf_s, hf16, encmask + t*TB,
          hfseq + (size_t)t*TB*THE, nullptr, nullptr, THE);
  }
  for (int t = TTIN-1; t >= 0; --t) {
    gemm(e16 + (size_t)t*TB*TE, TE, c16 + (size_t)t*TB*TF, TF, hb16, THE,
         nullptr, 0, WtBw, bbk, zbuf, TB, 4*THE);
    gates(zbuf, hb_s, cb_s, hb16, encmask + t*TB,
          hbseq + (size_t)t*TB*THE, nullptr, nullptr, THE);
  }

  // ===== decoder pre-net (big batched GEMMs, M = TOUT*B = 16384)
  k_drop16pad<<<((TTOUT*TB*TOP)+255)/256,256,0,stream>>>(y, dy16, 0x9E3779B9u);
  gemm(dy16, TOP, nullptr,0, nullptr,0, nullptr,0, WtP1, bp1, dp1, TTOUT*TB, TP);
  k_relu_drop16<<<((TTOUT*TB*TP)+255)/256,256,0,stream>>>(dp1, dp216, TTOUT*TB*TP, 0x85EBCA6Bu);
  gemm(dp216, TP, nullptr,0, nullptr,0, nullptr,0, WtP2, bp2, py2, TTOUT*TB, THE);
  k_relu_cvt<<<((TTOUT*TB*THE)+255)/256,256,0,stream>>>(py2, py216, TTOUT*TB*THE);

  // ===== seed decoder states from inputs (f32 + f16 shadow)
  hipMemcpyAsync(ah_s, d_in[2], (size_t)TB*THD*4,   hipMemcpyDeviceToDevice, stream);
  hipMemcpyAsync(ac_s, d_in[3], (size_t)TB*THD*4,   hipMemcpyDeviceToDevice, stream);
  hipMemcpyAsync(kk_s, d_in[4], (size_t)TB*TK*4,    hipMemcpyDeviceToDevice, stream);
  hipMemcpyAsync(w_s,  d_in[5], (size_t)TB*2*THE*4, hipMemcpyDeviceToDevice, stream);
  hipMemcpyAsync(h1_s, d_in[6], (size_t)TB*THD*4,   hipMemcpyDeviceToDevice, stream);
  hipMemcpyAsync(c1_s, d_in[7], (size_t)TB*THD*4,   hipMemcpyDeviceToDevice, stream);
  hipMemcpyAsync(h2_s, d_in[8], (size_t)TB*THD*4,   hipMemcpyDeviceToDevice, stream);
  hipMemcpyAsync(c2_s, d_in[9], (size_t)TB*THD*4,   hipMemcpyDeviceToDevice, stream);
  k_cvt<<<(TB*THD+255)/256,256,0,stream>>>(ah_s, ah16, TB*THD);
  k_cvt<<<(TB*2*THE+255)/256,256,0,stream>>>(w_s, w16, TB*2*THE);
  k_cvt<<<(TB*THD+255)/256,256,0,stream>>>(h1_s, h116, TB*THD);
  k_cvt<<<(TB*THD+255)/256,256,0,stream>>>(h2_s, h216, TB*THD);

  // ===== decoder scan: attention LSTM -> GMM attention -> LSTM1 -> LSTM2
  for (int t = 0; t < TTOUT; ++t) {
    const _Float16* q = py216 + (size_t)t*TB*THE;
    const float* mt = decmask + t*TB;
    // attention cell
    gemm(q, THE, w16, 2*THE, ah16, THD, nullptr, 0, WtAt, bat, zbuf, TB, 4*THD);
    gates(zbuf, ah_s, ac_s, ah16, mt, o_ah + (size_t)t*TB*THD,
          o_ac + (size_t)t*TB*THD, ahseq16 + (size_t)t*TB*THD, THD);
    // mixture params + kappa update
    k_abk<<<1, TB*3*TK, 0, stream>>>(ah_s, Wk, bk, kk_s, mt,
                                     o_ak + (size_t)t*TB*TK, a_sc, b_sc, kr_sc);
    // phi over encoder positions
    k_phi<<<(TB*TTIN+255)/256,256,0,stream>>>(a_sc, b_sc, kr_sc, encmask, mt,
                                              phib, o_aphi + (size_t)t*TB*TTIN);
    // context window
    k_ctx<<<(TB*2*THE+255)/256,256,0,stream>>>(phib, hfseq, hbseq, mt, w_s, w16,
                                               o_aw + (size_t)t*TB*2*THE,
                                               awseq16 + (size_t)t*TB*2*THE);
    // decoder LSTM 1
    gemm(q, THE, awseq16 + (size_t)t*TB*2*THE, 2*THE,
         ahseq16 + (size_t)t*TB*THD, THD, h116, THD, Wt1, b1, zbuf, TB, 4*THD);
    gates(zbuf, h1_s, c1_s, h116, mt, o_h1 + (size_t)t*TB*THD,
          o_c1 + (size_t)t*TB*THD, h1seq16 + (size_t)t*TB*THD, THD);
    // decoder LSTM 2
    gemm(q, THE, awseq16 + (size_t)t*TB*2*THE, 2*THE,
         h1seq16 + (size_t)t*TB*THD, THD, h216, THD, Wt2, b2, zbuf, TB, 4*THD);
    gates(zbuf, h2_s, c2_s, h216, mt, o_h2 + (size_t)t*TB*THD,
          o_c2 + (size_t)t*TB*THD, h2seq16 + (size_t)t*TB*THD, THD);
  }

  // ===== output projection (M = 16384, K = 512, N = 80)
  gemm(h2seq16, THD, nullptr,0, nullptr,0, nullptr,0, WtO, bo, o_preds, TTOUT*TB, TO);
}
